// VariationalGRU_13537736917055
// MI455X (gfx1250) — matrix operand order repaired and checked by reading, not compiled, not run on hardware
//
#include <hip/hip_runtime.h>
#include <hip/hip_bf16.h>

// Variational GRU on MI455X (gfx1250): persistent 16-WG kernel, bf16 WMMA
// (v_wmma_f32_16x16x32_bf16), global-barrier recurrence, weights prepacked
// into WMMA-B fragment layout in workspace (lives in L2).
// Round 4: 2-way ping-pong unroll inside the rolled kc loops (overlap L2
// fragment loads with WMMA, partial loadcnt waits) and fast
// v_exp/v_rcp-based sigmoid/tanh epilogue (no IEEE div chains).

#define B_DIM 256
#define L_DIM 512
#define C_DIM 128
#define H_DIM 256
#define NWG   16
#define NTHREADS 512

typedef __attribute__((ext_vector_type(8)))  float  v8f;
typedef __attribute__((ext_vector_type(16))) __bf16 v16bf;

union FragU { uint4 u[2]; v16bf v; };
union Pk4 { unsigned short s[4]; uint2 u; };

__device__ __forceinline__ unsigned short f2bf(float x) {
  unsigned u = __float_as_uint(x);
  return (unsigned short)((u + 0x7FFFu + ((u >> 16) & 1u)) >> 16);
}

// One hash -> 4 dropout mask bytes for columns grp*4 .. grp*4+3.
__device__ __forceinline__ unsigned hash32(unsigned t, unsigned kind,
                                           unsigned row, unsigned grp) {
  unsigned x = (t * 3u + kind + 1u) * 0x9E3779B1u;
  x ^= row * 0x85EBCA77u;
  x ^= grp * 0xC2B2AE3Du;
  x ^= x >> 16; x *= 0x7FEB352Du;
  x ^= x >> 15; x *= 0x846CA68Bu;
  x ^= x >> 16;
  return x;
}
// keep w.p. 205/256 ~= 0.8 ; scale 1/(1-0.2)
__device__ __forceinline__ float maskof(unsigned h, int sub) {
  return ((h >> (sub * 8)) & 0xFFu) >= 51u ? 1.25f : 0.0f;
}

// Fast gates: v_exp_f32 + v_rcp_f32 (no IEEE division sequences).
__device__ __forceinline__ float fast_sigmoid(float x) {
  return __builtin_amdgcn_rcpf(1.f + __expf(-x));
}
__device__ __forceinline__ float fast_tanh(float x) {
  return 1.f - 2.f * __builtin_amdgcn_rcpf(1.f + __expf(2.f * x));
}

__device__ __forceinline__ v8f wmma_bf16(v16bf a, v16bf b, v8f c) {
  return __builtin_amdgcn_wmma_f32_16x16x32_bf16(
      false, a, false, b, (short)0, c, false, false);
}

// B fragment (32Kx16N bf16): lane L owns 16 contiguous bf16 (32B).
__device__ __forceinline__ v16bf load_bfrag_p(const unsigned short* p) {
  const uint4* q = reinterpret_cast<const uint4*>(p);
  FragU f; f.u[0] = q[0]; f.u[1] = q[1];
  return f.v;
}

// A fragment (16Mx32K bf16) from LDS at a running per-lane pointer.
__device__ __forceinline__ v16bf load_afrag_p(const unsigned short* p) {
  const uint4* q = reinterpret_cast<const uint4*>(p);
  FragU f; f.u[0] = q[0]; f.u[1] = q[1];  // 16 contiguous bf16
  return f.v;
}

__device__ __forceinline__ void grid_barrier(unsigned* cnt, unsigned target) {
  __syncthreads();
  if (threadIdx.x == 0) {
    __threadfence();
    atomicAdd(cnt, 1u);
    while (__hip_atomic_load(cnt, __ATOMIC_RELAXED, __HIP_MEMORY_SCOPE_AGENT) < target)
      __builtin_amdgcn_s_sleep(2);
    __threadfence();
  }
  __syncthreads();
}

// ---------------- prep kernels ----------------

__global__ void init_state(const float* __restrict__ h0, float* __restrict__ hbuf,
                           unsigned* __restrict__ cnt, int n) {
  int i = blockIdx.x * blockDim.x + threadIdx.x;
  if (i < n) hbuf[i] = h0[i];
  if (i == 0) *cnt = 0;
}

__global__ void pack_weights(const float* __restrict__ W, unsigned short* __restrict__ dst,
                             int K, int kchunks, int total) {
  int idx = blockIdx.x * blockDim.x + threadIdx.x;
  if (idx >= total) return;
  int r = idx / K, k = idx - r * K;
  int lane = (((k >> 4) & 1) << 4) | (r & 15);
  size_t di = ((size_t)(r >> 4) * kchunks + (k >> 5)) * 512 + lane * 16 + (k & 15);
  dst[di] = f2bf(W[idx]);
}

// ---------------- main persistent kernel ----------------

__global__ __launch_bounds__(NTHREADS, 1)
void gru_persistent(const float* __restrict__ x,
                    const float* __restrict__ b_ih,
                    const float* __restrict__ b_hh,
                    const unsigned short* __restrict__ whhP,
                    const unsigned short* __restrict__ wihP,
                    float* __restrict__ hbuf,
                    unsigned* __restrict__ cnt,
                    float* __restrict__ out) {
  constexpr int HSTR = H_DIM + 8;
  constexpr int XSTR = C_DIM + 8;
  __shared__ __align__(16) unsigned short h_ld[64 * HSTR];  // 33792 B
  __shared__ __align__(16) unsigned short x_ld[64 * XSTR];  // 17408 B

  const int wg = blockIdx.x;
  const int bib = (wg >> 2) * 64;       // B-tile base
  const int hjb = (wg & 3) * 64;        // H-tile base
  const int tid = threadIdx.x;
  const int wave = tid >> 5, lane = tid & 31;
  const int mt = wave & 3;              // 16-row M tile
  const int nt = wave >> 2;             // 16-col N tile
  const int lanelo = lane & 15, lanehi = lane >> 4;

  // Per-lane constants hoisted out of the recurrence.
  const int rb0 = hjb + nt * 16;        // gate-row base of this wave's tile
  const int j0 = rb0 + lanelo;          // hidden column this lane accumulates
  const float bR  = b_ih[j0] + b_hh[j0];
  const float bZ  = b_ih[H_DIM + j0] + b_hh[H_DIM + j0];
  const float bNi = b_ih[2 * H_DIM + j0];
  const float bNh = b_hh[2 * H_DIM + j0];
  const int brow = bib + mt * 16 + (lanehi << 3);  // batch row of acc elem 0

  // Running-pointer bases (per lane), advanced per k-chunk.
  const int laneoff = lane * 16;
  const unsigned short* whh_R0 = whhP + ((size_t)(rb0 >> 4) * 8) * 512 + laneoff;
  const unsigned short* whh_Z0 = whhP + ((size_t)((H_DIM + rb0) >> 4) * 8) * 512 + laneoff;
  const unsigned short* whh_N0 = whhP + ((size_t)((2 * H_DIM + rb0) >> 4) * 8) * 512 + laneoff;
  const unsigned short* wih_R0 = wihP + ((size_t)(rb0 >> 4) * 4) * 512 + laneoff;
  const unsigned short* wih_Z0 = wihP + ((size_t)((H_DIM + rb0) >> 4) * 4) * 512 + laneoff;
  const unsigned short* wih_N0 = wihP + ((size_t)((2 * H_DIM + rb0) >> 4) * 4) * 512 + laneoff;
  const unsigned short* aH0 = &h_ld[(mt * 16 + lanelo) * HSTR + (lanehi << 4)];
  const unsigned short* aX0 = &x_ld[(mt * 16 + lanelo) * XSTR + (lanehi << 4)];

  float* out_last = out + (size_t)B_DIM * L_DIM * H_DIM;

  for (int t = 0; t < L_DIM; ++t) {
    const float* hprev = hbuf + (size_t)(t & 1) * B_DIM * H_DIM;
    float*       hnext = hbuf + (size_t)((t + 1) & 1) * B_DIM * H_DIM;

    // ---- stage h_d (recurrent dropout, kind 1), 4 cols per iteration ----
    for (int i = tid; i < 64 * (H_DIM / 4); i += NTHREADS) {
      int m = i >> 6, g = i & 63;
      int b = bib + m, k0 = g * 4;
      float4 v = *reinterpret_cast<const float4*>(&hprev[b * H_DIM + k0]);
      unsigned h = hash32(t, 1, b, g);
      Pk4 pk;
      pk.s[0] = f2bf(v.x * maskof(h, 0));
      pk.s[1] = f2bf(v.y * maskof(h, 1));
      pk.s[2] = f2bf(v.z * maskof(h, 2));
      pk.s[3] = f2bf(v.w * maskof(h, 3));
      *reinterpret_cast<uint2*>(&h_ld[m * HSTR + k0]) = pk.u;
    }
    // ---- stage x_t (input dropout, kind 0) ----
    for (int i = tid; i < 64 * (C_DIM / 4); i += NTHREADS) {
      int m = i >> 5, g = i & 31;
      int b = bib + m, c0 = g * 4;
      float4 v = *reinterpret_cast<const float4*>(
          &x[((size_t)b * L_DIM + t) * C_DIM + c0]);
      unsigned h = hash32(t, 0, b, g);
      Pk4 pk;
      pk.s[0] = f2bf(v.x * maskof(h, 0));
      pk.s[1] = f2bf(v.y * maskof(h, 1));
      pk.s[2] = f2bf(v.z * maskof(h, 2));
      pk.s[3] = f2bf(v.w * maskof(h, 3));
      *reinterpret_cast<uint2*>(&x_ld[m * XSTR + c0]) = pk.u;
    }
    __syncthreads();

    // ---- accumulators (R/Z fold gi+gh+biases; N split: n = tanh(Ni + r*Nh))
    v8f accR, accZ, accNi, accNh;
#pragma unroll
    for (int v = 0; v < 8; ++v) {
      accR[v] = bR; accZ[v] = bZ; accNi[v] = bNi; accNh[v] = bNh;
    }

    // gh = h_d @ W_hh^T (K=256, 8 chunks).  Rolled loop, 2 chunks per
    // iteration: ~8 loads in flight before the first WMMA waits, while
    // live registers stay ~130 (no spills, no whole-stream hoisting).
    {
      const unsigned short* pa = aH0;
      const unsigned short* pr = whh_R0;
      const unsigned short* pz = whh_Z0;
      const unsigned short* pn = whh_N0;
#pragma unroll 1
      for (int kc = 0; kc < 4; ++kc) {
        v16bf a0 = load_afrag_p(pa);
        v16bf r0 = load_bfrag_p(pr);
        v16bf z0 = load_bfrag_p(pz);
        v16bf n0 = load_bfrag_p(pn);
        v16bf a1 = load_afrag_p(pa + 32);
        v16bf r1 = load_bfrag_p(pr + 512);
        v16bf z1 = load_bfrag_p(pz + 512);
        v16bf n1 = load_bfrag_p(pn + 512);
        accR  = wmma_bf16(a0, r0, accR);
        accZ  = wmma_bf16(a0, z0, accZ);
        accNh = wmma_bf16(a0, n0, accNh);
        accR  = wmma_bf16(a1, r1, accR);
        accZ  = wmma_bf16(a1, z1, accZ);
        accNh = wmma_bf16(a1, n1, accNh);
        pa += 64; pr += 1024; pz += 1024; pn += 1024;
      }
    }
    // gi = x_t @ W_ih^T (K=128, 4 chunks), same 2-way ping-pong.
    {
      const unsigned short* pa = aX0;
      const unsigned short* pr = wih_R0;
      const unsigned short* pz = wih_Z0;
      const unsigned short* pn = wih_N0;
#pragma unroll 1
      for (int kc = 0; kc < 2; ++kc) {
        v16bf a0 = load_afrag_p(pa);
        v16bf r0 = load_bfrag_p(pr);
        v16bf z0 = load_bfrag_p(pz);
        v16bf n0 = load_bfrag_p(pn);
        v16bf a1 = load_afrag_p(pa + 32);
        v16bf r1 = load_bfrag_p(pr + 512);
        v16bf z1 = load_bfrag_p(pz + 512);
        v16bf n1 = load_bfrag_p(pn + 512);
        accR  = wmma_bf16(a0, r0, accR);
        accZ  = wmma_bf16(a0, z0, accZ);
        accNi = wmma_bf16(a0, n0, accNi);
        accR  = wmma_bf16(a1, r1, accR);
        accZ  = wmma_bf16(a1, z1, accZ);
        accNi = wmma_bf16(a1, n1, accNi);
        pa += 64; pr += 1024; pz += 1024; pn += 1024;
      }
    }

    // ---- epilogue: gates, GRU update, output dropout ----
    // C/D layout: VGPR v, lane -> M = v + (lane>=16 ? 8 : 0), N = lane&15.
    const unsigned hgrp = (unsigned)(j0 >> 2);
    const int hsub = j0 & 3;
#pragma unroll
    for (int v = 0; v < 8; ++v) {
      int b = brow + v;
      float rr = fast_sigmoid(accR[v]);
      float zz = fast_sigmoid(accZ[v]);
      float nn = fast_tanh(accNi[v] + rr * accNh[v]);
      float hd = hprev[b * H_DIM + j0] * maskof(hash32(t, 1, b, hgrp), hsub);
      float ho = ((1.f - zz) * nn + zz * hd) * maskof(hash32(t, 2, b, hgrp), hsub);
      out[((size_t)b * L_DIM + t) * H_DIM + j0] = ho;
      hnext[b * H_DIM + j0] = ho;
      if (t == L_DIM - 1) out_last[(size_t)b * H_DIM + j0] = ho;
    }

    grid_barrier(cnt, (unsigned)(t + 1) * NWG);
  }
}

extern "C" void kernel_launch(void* const* d_in, const int* in_sizes, int n_in,
                              void* d_out, int out_size, void* d_ws, size_t ws_size,
                              hipStream_t stream) {
  (void)in_sizes; (void)n_in; (void)out_size; (void)ws_size;
  const float* x   = (const float*)d_in[0];
  const float* h0  = (const float*)d_in[1];   // [1, B, H]
  const float* Wih = (const float*)d_in[2];   // [768, 128]
  const float* Whh = (const float*)d_in[3];   // [768, 256]
  const float* bih = (const float*)d_in[4];
  const float* bhh = (const float*)d_in[5];
  float* out = (float*)d_out;

  // ws: [0,1024) barrier counter | h double-buffer f32 | bf16 W_hh | bf16 W_ih
  char* base = (char*)d_ws;
  unsigned* cnt = (unsigned*)base;
  float* hbuf = (float*)(base + 1024);
  unsigned short* whhP =
      (unsigned short*)(base + 1024 + (size_t)2 * B_DIM * H_DIM * sizeof(float));
  unsigned short* wihP = whhP + 3 * H_DIM * H_DIM;

  init_state<<<(B_DIM * H_DIM + 255) / 256, 256, 0, stream>>>(
      h0, hbuf, cnt, B_DIM * H_DIM);
  pack_weights<<<(3 * H_DIM * H_DIM + 255) / 256, 256, 0, stream>>>(
      Whh, whhP, H_DIM, 8, 3 * H_DIM * H_DIM);
  pack_weights<<<(3 * H_DIM * C_DIM + 255) / 256, 256, 0, stream>>>(
      Wih, wihP, C_DIM, 4, 3 * H_DIM * C_DIM);
  gru_persistent<<<NWG, NTHREADS, 0, stream>>>(
      x, bih, bhh, whhP, wihP, hbuf, cnt, out);
}
